// SelectiveScanLayer_88467736363292
// MI455X (gfx1250) — compile-verified, matrix operand order checked by reading, not run on hardware
//
#include <hip/hip_runtime.h>
#include <hip/hip_bf16.h>

// ---------------------------------------------------------------------------
// Shapes (fixed by the reference)
// ---------------------------------------------------------------------------
#define BB   4
#define LL   4096
#define MM   1024
#define SS   64
#define ROWS (BB * LL)          // 16384 tokens
#define NC   16                 // scan chunks
#define LC   (LL / NC)          // 256 per chunk

typedef __attribute__((ext_vector_type(16))) _Float16 v16h;
typedef __attribute__((ext_vector_type(8)))  _Float16 v8h;
typedef __attribute__((ext_vector_type(4)))  _Float16 v4h;
typedef __attribute__((ext_vector_type(8)))  float    v8f;

// ---------------------------------------------------------------------------
// WMMA helpers: CDNA5 16x16x32 f16 fragments.
// A (16x32, MxK) / B (32x16, KxN) lane layout (ISA 7.12.2):
//   lane%16 = M (or N) index; lane/16 selects K-half.
//   v16h elems [0..7]  = K { khalf*8 + 0..7 }      (16 contiguous bytes)
//   v16h elems [8..15] = K { 16 + khalf*8 + 0..7 } (16 contiguous bytes)
// Both A and B here are row-major [rows, K], so one loader serves both.
// ---------------------------------------------------------------------------
__device__ __forceinline__ v16h load_frag(const _Float16* __restrict__ p) {
    v8h lo = *(const v8h*)(p);        // K = khalf*8 + 0..7
    v8h hi = *(const v8h*)(p + 16);   // K = 16 + khalf*8 + 0..7
    v16h t;
#pragma unroll
    for (int i = 0; i < 8; ++i) { t[i] = lo[i]; t[i + 8] = hi[i]; }
    return t;
}

struct Acc44 { v8f c[4][4]; };

// One wave computes a 64x64 C tile: 4x4 fragments, K stepped by 32.
__device__ __forceinline__ void wave_gemm64(const _Float16* __restrict__ A,
                                            const _Float16* __restrict__ B,
                                            int ldA, int ldB,
                                            int rowbase, int colbase, int K,
                                            Acc44& acc) {
    const int lane  = threadIdx.x & 31;
    const int rr    = lane & 15;
    const int khalf = (lane >> 4) & 1;
    const _Float16* Ab[4];
    const _Float16* Bb[4];
#pragma unroll
    for (int t = 0; t < 4; ++t) {
        Ab[t] = A + (size_t)(rowbase + t * 16 + rr) * ldA + khalf * 8;
        Bb[t] = B + (size_t)(colbase + t * 16 + rr) * ldB + khalf * 8;
    }
    for (int kb = 0; kb < K; kb += 32) {
        v16h a[4], b[4];
#pragma unroll
        for (int t = 0; t < 4; ++t) a[t] = load_frag(Ab[t] + kb);
#pragma unroll
        for (int t = 0; t < 4; ++t) b[t] = load_frag(Bb[t] + kb);
#pragma unroll
        for (int i = 0; i < 4; ++i)
#pragma unroll
            for (int j = 0; j < 4; ++j)
                acc.c[i][j] = __builtin_amdgcn_wmma_f32_16x16x32_f16(
                    false, a[i], false, b[j], (short)0, acc.c[i][j], false, false);
    }
}

// sigmoid via v_exp_f32 + v_rcp_f32 (avoids the full IEEE divide expansion)
__device__ __forceinline__ float fast_sigmoid(float v) {
    return __builtin_amdgcn_rcpf(1.0f + __expf(-v));
}

// ---------------------------------------------------------------------------
// Stage 0: fp32 -> fp16 weight conversion (Dmat transposed to [M,S])
// ---------------------------------------------------------------------------
__global__ void k_convert(const float* __restrict__ W_in,
                          const float* __restrict__ W_xp,
                          const float* __restrict__ W_out,
                          const float* __restrict__ Dmat,
                          _Float16* __restrict__ Wih,
                          _Float16* __restrict__ Wxh,
                          _Float16* __restrict__ Woh,
                          _Float16* __restrict__ Dth) {
    const int NWI = 2 * MM * MM, NXP = SS * MM, NWO = MM * MM, ND = SS * MM;
    int i = blockIdx.x * 256 + threadIdx.x;
    if (i < NWI) { Wih[i] = (_Float16)W_in[i]; return; }
    i -= NWI;
    if (i < NXP) { Wxh[i] = (_Float16)W_xp[i]; return; }
    i -= NXP;
    if (i < NWO) { Woh[i] = (_Float16)W_out[i]; return; }
    i -= NWO;
    if (i < ND) {
        int n = i >> 6, s = i & 63;                 // Dt[n,s] = Dmat[s,n]
        Dth[i] = (_Float16)Dmat[(size_t)s * MM + n];
    }
}

// ---------------------------------------------------------------------------
// Stage 1: LayerNorm (one 256-thread block per token), fp16 output
// ---------------------------------------------------------------------------
__global__ __launch_bounds__(256) void k_layernorm(const float* __restrict__ x,
                                                   const float* __restrict__ gamma,
                                                   const float* __restrict__ beta,
                                                   _Float16* __restrict__ xn) {
    __shared__ float sSum[8], sSq[8];
    const int row = blockIdx.x;
    const size_t base = (size_t)row * MM;
    const int tid = threadIdx.x;

    float4 v = ((const float4*)(x + base))[tid];
    float s = v.x + v.y + v.z + v.w;
    float q = v.x * v.x + v.y * v.y + v.z * v.z + v.w * v.w;
#pragma unroll
    for (int off = 16; off > 0; off >>= 1) {
        s += __shfl_down(s, off);
        q += __shfl_down(q, off);
    }
    if ((tid & 31) == 0) { sSum[tid >> 5] = s; sSq[tid >> 5] = q; }
    __syncthreads();
    float ts = 0.f, tq = 0.f;
#pragma unroll
    for (int i = 0; i < 8; ++i) { ts += sSum[i]; tq += sSq[i]; }
    const float mu = ts * (1.0f / MM);
    const float var = tq * (1.0f / MM) - mu * mu;
    const float rstd = rsqrtf(var + 1e-5f);

    float4 g = ((const float4*)gamma)[tid];
    float4 bb = ((const float4*)beta)[tid];
    v4h o;
    o[0] = (_Float16)((v.x - mu) * rstd * g.x + bb.x);
    o[1] = (_Float16)((v.y - mu) * rstd * g.y + bb.y);
    o[2] = (_Float16)((v.z - mu) * rstd * g.z + bb.z);
    o[3] = (_Float16)((v.w - mu) * rstd * g.w + bb.w);
    ((v4h*)(xn + base))[tid] = o;
}

// ---------------------------------------------------------------------------
// Stage 2: GEMM1  xz = xn @ W_in^T + b_in ; fused SiLU / sigmoid split
//   block = 8 waves in 2x4 -> 128x256 C tile ; grid (128, 8)
// ---------------------------------------------------------------------------
__global__ __launch_bounds__(256) void k_gemm1(const _Float16* __restrict__ xn,
                                               const _Float16* __restrict__ Wih,
                                               const float* __restrict__ b_in,
                                               _Float16* __restrict__ xp,
                                               _Float16* __restrict__ zg) {
    Acc44 acc = {};
    const int w = threadIdx.x >> 5;
    const int rowbase = blockIdx.x * 128 + (w & 1) * 64;
    const int colbase = blockIdx.y * 256 + (w >> 1) * 64;
    wave_gemm64(xn, Wih, MM, MM, rowbase, colbase, MM, acc);

    const int lane = threadIdx.x & 31;
    const int nlo = lane & 15, mh = (lane >> 4) * 8;
    const bool is_x = (blockIdx.y < (MM / 256));     // uniform: SiLU half vs gate half
#pragma unroll
    for (int j = 0; j < 4; ++j) {
        const int n = colbase + j * 16 + nlo;
        const float bias = b_in[n];
#pragma unroll
        for (int i = 0; i < 4; ++i) {
#pragma unroll
            for (int jj = 0; jj < 8; ++jj) {
                const int m = rowbase + i * 16 + mh + jj;
                const float v = acc.c[i][j][jj] + bias;
                const float sg = fast_sigmoid(v);
                if (is_x) xp[(size_t)m * MM + n] = (_Float16)(v * sg);          // SiLU
                else      zg[(size_t)m * MM + (n - MM)] = (_Float16)sg;         // gate
            }
        }
    }
}

// ---------------------------------------------------------------------------
// Stage 3: GEMM2  u = x_p @ W_xp^T + b_xp   (cols = 64)
//   block = 8 waves stacked in rows -> 512x64 ; grid (32, 1)
// ---------------------------------------------------------------------------
__global__ __launch_bounds__(256) void k_gemm2(const _Float16* __restrict__ xp,
                                               const _Float16* __restrict__ Wxh,
                                               const float* __restrict__ b_xp,
                                               float* __restrict__ u) {
    Acc44 acc = {};
    const int w = threadIdx.x >> 5;
    const int rowbase = blockIdx.x * 512 + w * 64;
    wave_gemm64(xp, Wxh, MM, MM, rowbase, 0, MM, acc);

    const int lane = threadIdx.x & 31;
    const int nlo = lane & 15, mh = (lane >> 4) * 8;
#pragma unroll
    for (int j = 0; j < 4; ++j) {
        const int n = j * 16 + nlo;
        const float bias = b_xp[n];
#pragma unroll
        for (int i = 0; i < 4; ++i)
#pragma unroll
            for (int jj = 0; jj < 8; ++jj) {
                const int m = rowbase + i * 16 + mh + jj;
                u[(size_t)m * SS + n] = acc.c[i][j][jj] + bias;
            }
    }
}

// ---------------------------------------------------------------------------
// Stage 4: chunked parallel scan  h[l] = a*h[l-1] + u[l]
// ---------------------------------------------------------------------------
__global__ void k_scan_chunk(const float* __restrict__ u,
                             const float* __restrict__ A_log,
                             float* __restrict__ hloc,
                             float* __restrict__ chunk_end) {
    const int t = blockIdx.x * 256 + threadIdx.x;   // B*S*NC = 4096 threads
    const int s = t & 63;
    const int c = (t >> 6) & (NC - 1);
    const int b = t >> 10;
    const float a = expf(-expf(A_log[s]));
    float h = 0.f;
    const int l0 = c * LC;
    for (int j = 0; j < LC; ++j) {
        const size_t idx = ((size_t)b * LL + l0 + j) * SS + s;
        h = a * h + u[idx];
        hloc[idx] = h;
    }
    chunk_end[((b * SS + s) * NC) + c] = h;
}

__global__ void k_scan_carry(const float* __restrict__ A_log,
                             const float* __restrict__ chunk_end,
                             float* __restrict__ carry_in) {
    const int t = threadIdx.x;                      // 256 = B*S
    const int s = t & 63, b = t >> 6;
    const float E = expf(-expf(A_log[s]) * (float)LC);  // a^LC
    float carry = 0.f;
    for (int c = 0; c < NC; ++c) {
        carry_in[((b * SS + s) * NC) + c] = carry;
        carry = E * carry + chunk_end[((b * SS + s) * NC) + c];
    }
}

// apply carries, emit fp16 h in [l*B + b, s] order (the [L,B,M] order the
// tail GEMMs need, which also realizes the reference's reshape trick)
__global__ void k_scan_apply(const float* __restrict__ hloc,
                             const float* __restrict__ A_log,
                             const float* __restrict__ carry_in,
                             _Float16* __restrict__ h_h) {
    const int t = blockIdx.x * 256 + threadIdx.x;
    const int s = t & 63;
    const int c = (t >> 6) & (NC - 1);
    const int b = t >> 10;
    const float a = expf(-expf(A_log[s]));
    const float carry = carry_in[((b * SS + s) * NC) + c];
    float p = a;
    const int l0 = c * LC;
    for (int j = 0; j < LC; ++j) {
        const int l = l0 + j;
        const float hv = hloc[((size_t)b * LL + l) * SS + s] + carry * p;
        p *= a;
        h_h[((size_t)l * BB + b) * SS + s] = (_Float16)hv;
    }
}

// ---------------------------------------------------------------------------
// Stage 5: GEMM3  ys = (h @ Dmat) * z   (K = 64, fused gating, fp16 out)
//   rows indexed r' = l*B + b ; grid (128, 4)
// ---------------------------------------------------------------------------
__global__ __launch_bounds__(256) void k_gemm3(const _Float16* __restrict__ h_h,
                                               const _Float16* __restrict__ Dth,
                                               const _Float16* __restrict__ zg,
                                               _Float16* __restrict__ ys) {
    Acc44 acc = {};
    const int w = threadIdx.x >> 5;
    const int rowbase = blockIdx.x * 128 + (w & 1) * 64;
    const int colbase = blockIdx.y * 256 + (w >> 1) * 64;
    wave_gemm64(h_h, Dth, SS, SS, rowbase, colbase, SS, acc);

    const int lane = threadIdx.x & 31;
    const int nlo = lane & 15, mh = (lane >> 4) * 8;
#pragma unroll
    for (int i = 0; i < 4; ++i)
#pragma unroll
        for (int j = 0; j < 4; ++j)
#pragma unroll
            for (int jj = 0; jj < 8; ++jj) {
                const int rp = rowbase + i * 16 + mh + jj;   // r' = l*B + b
                const int n = colbase + j * 16 + nlo;
                const int b = rp & (BB - 1), l = rp >> 2;
                const float z = (float)zg[((size_t)b * LL + l) * MM + n];
                ys[(size_t)rp * MM + n] = (_Float16)(acc.c[i][j][jj] * z);
            }
}

// ---------------------------------------------------------------------------
// Stage 6: GEMM4  out = x + (ys @ W_out^T + b_out)   ; grid (128, 4)
// ---------------------------------------------------------------------------
__global__ __launch_bounds__(256) void k_gemm4(const _Float16* __restrict__ ys,
                                               const _Float16* __restrict__ Woh,
                                               const float* __restrict__ b_out,
                                               const float* __restrict__ xflat,
                                               float* __restrict__ out) {
    Acc44 acc = {};
    const int w = threadIdx.x >> 5;
    const int rowbase = blockIdx.x * 128 + (w & 1) * 64;
    const int colbase = blockIdx.y * 256 + (w >> 1) * 64;
    wave_gemm64(ys, Woh, MM, MM, rowbase, colbase, MM, acc);

    const int lane = threadIdx.x & 31;
    const int nlo = lane & 15, mh = (lane >> 4) * 8;
#pragma unroll
    for (int j = 0; j < 4; ++j) {
        const int n = colbase + j * 16 + nlo;
        const float bias = b_out[n];
#pragma unroll
        for (int i = 0; i < 4; ++i)
#pragma unroll
            for (int jj = 0; jj < 8; ++jj) {
                const int m = rowbase + i * 16 + mh + jj;
                const size_t idx = (size_t)m * MM + n;   // y[L,B,M].flat == x[B,L,M].flat
                out[idx] = xflat[idx] + acc.c[i][j][jj] + bias;
            }
    }
}

// ---------------------------------------------------------------------------
// Host-side launch
// ---------------------------------------------------------------------------
extern "C" void kernel_launch(void* const* d_in, const int* in_sizes, int n_in,
                              void* d_out, int out_size, void* d_ws, size_t ws_size,
                              hipStream_t stream) {
    const float* x     = (const float*)d_in[0];
    const float* A_log = (const float*)d_in[1];
    const float* Dmat  = (const float*)d_in[2];
    const float* W_in  = (const float*)d_in[3];
    const float* b_in  = (const float*)d_in[4];
    const float* W_xp  = (const float*)d_in[5];
    const float* b_xp  = (const float*)d_in[6];
    const float* W_out = (const float*)d_in[7];
    const float* b_out = (const float*)d_in[8];
    const float* gamma = (const float*)d_in[9];
    const float* beta  = (const float*)d_in[10];
    float* out = (float*)d_out;

    // workspace carve-up (all sizes are multiples of 4KB)
    char* ws = (char*)d_ws;
    size_t off = 0;
    auto take = [&](size_t bytes) { char* p = ws + off; off += bytes; return p; };
    _Float16* xn_h  = (_Float16*)take((size_t)ROWS * MM * 2);   // 32 MB
    _Float16* xp_h  = (_Float16*)take((size_t)ROWS * MM * 2);   // 32 MB
    _Float16* z_h   = (_Float16*)take((size_t)ROWS * MM * 2);   // 32 MB
    _Float16* ys_h  = (_Float16*)take((size_t)ROWS * MM * 2);   // 32 MB
    float*    u     = (float*)take((size_t)ROWS * SS * 4);      //  4 MB
    float*    hloc  = (float*)take((size_t)ROWS * SS * 4);      //  4 MB
    _Float16* h_h   = (_Float16*)take((size_t)ROWS * SS * 2);   //  2 MB
    _Float16* Wih   = (_Float16*)take((size_t)2 * MM * MM * 2); //  4 MB
    _Float16* Wxh   = (_Float16*)take((size_t)SS * MM * 2);
    _Float16* Woh   = (_Float16*)take((size_t)MM * MM * 2);     //  2 MB
    _Float16* Dth   = (_Float16*)take((size_t)SS * MM * 2);
    float*    cend  = (float*)take((size_t)BB * SS * NC * 4);
    float*    cin   = (float*)take((size_t)BB * SS * NC * 4);
    (void)ws_size; (void)in_sizes; (void)n_in; (void)out_size;

    // 0) weight conversion
    {
        const int total = 2 * MM * MM + SS * MM + MM * MM + SS * MM;
        k_convert<<<(total + 255) / 256, 256, 0, stream>>>(W_in, W_xp, W_out, Dmat,
                                                           Wih, Wxh, Woh, Dth);
    }
    // 1) layernorm
    k_layernorm<<<ROWS, 256, 0, stream>>>(x, gamma, beta, xn_h);
    // 2) in_proj + gates
    k_gemm1<<<dim3(ROWS / 128, (2 * MM) / 256), 256, 0, stream>>>(xn_h, Wih, b_in, xp_h, z_h);
    // 3) state projection
    k_gemm2<<<dim3(ROWS / 512, 1), 256, 0, stream>>>(xp_h, Wxh, b_xp, u);
    // 4) parallel scan
    k_scan_chunk<<<(BB * SS * NC) / 256, 256, 0, stream>>>(u, A_log, hloc, cend);
    k_scan_carry<<<1, BB * SS, 0, stream>>>(A_log, cend, cin);
    k_scan_apply<<<(BB * SS * NC) / 256, 256, 0, stream>>>(hloc, A_log, cin, h_h);
    // 5) readout + gating
    k_gemm3<<<dim3(ROWS / 128, MM / 256), 256, 0, stream>>>(h_h, Dth, z_h, ys_h);
    // 6) out_proj + residual
    k_gemm4<<<dim3(ROWS / 128, MM / 256), 256, 0, stream>>>(ys_h, Woh, b_out, x, out);
}
